// SparseMoELayer_59682865545680
// MI455X (gfx1250) — matrix operand order repaired and checked by reading, not compiled, not run on hardware
//
#include <hip/hip_runtime.h>
#include <hip/hip_bf16.h>
#include <math.h>

#define D_MODEL 1024
#define D_FF    4096
#define NE      8
#define NTOK    8192
#define NASSIGN (NTOK * 2)
#define BM      64   // tokens per FFN block
#define KC      128  // D_FF chunk width

typedef __attribute__((ext_vector_type(16))) __bf16 v16bf;
typedef __attribute__((ext_vector_type(8)))  float  v8f;

union AFrag { v16bf v; uint4 q[2]; };

// ---------------------------------------------------------------- helpers

__device__ __forceinline__ unsigned pack2_bf16_rne(float a, float b) {
  unsigned ua = __float_as_uint(a);
  unsigned ub = __float_as_uint(b);
  ua = (ua + 0x7fffu + ((ua >> 16) & 1u)) >> 16;
  ub = (ub + 0x7fffu + ((ub >> 16) & 1u)) >> 16;
  return (ua & 0xffffu) | (ub << 16);
}

// A fragment (16x32 bf16, MxK) from row-major LDS.
// ISA 7.12.2: lanes 0-15/16-31 hold M=0..15; VGPR0-3 K={0..7}+8*half,
// VGPR4-7 K={16..23}+8*half -> two 16B reads per lane.
__device__ __forceinline__ v16bf load_a_frag(const __bf16* base, int stride,
                                             int row0, int kbase) {
  const int lane = threadIdx.x & 31;
  const int m    = lane & 15;
  const int half = lane >> 4;
  const __bf16* p = base + (row0 + m) * stride + kbase + half * 8;
  AFrag f;
  f.q[0] = *(const uint4*)(p);
  f.q[1] = *(const uint4*)(p + 16);
  return f.v;
}

__device__ __forceinline__ float block_reduce_sum_256(float v, float* red) {
  const int tid = threadIdx.x;
  red[tid] = v;
  __syncthreads();
#pragma unroll
  for (int off = 128; off > 0; off >>= 1) {
    if (tid < off) red[tid] += red[tid + off];
    __syncthreads();
  }
  float r = red[0];
  __syncthreads();
  return r;
}

// ---------------------------------------------------------------- weight swizzle
// Converts fp32 [NE][K][N] row-major weights into bf16 fragments laid out as
// [e][n_tile][k_tile][lane][8 dwords] (ISA 7.12.5 B layout), so the GEMM loop
// streams B fragments with contiguous 128-bit loads and zero conversion VALU.

__global__ __launch_bounds__(256)
void swizzle_w_kernel(const float* __restrict__ W, unsigned* __restrict__ S,
                      int K, int N) {
  const long idx = (long)blockIdx.x * 256 + threadIdx.x;   // dword index
  const int v2   = (int)(idx & 7);
  const int lane = (int)((idx >> 3) & 31);
  const long tile = idx >> 8;
  const int nkt = K >> 5;
  const int nnt = N >> 4;
  const int kt = (int)(tile % nkt);
  const long t2 = tile / nkt;
  const int nt = (int)(t2 % nnt);
  const int e  = (int)(t2 / nnt);
  if (e >= NE) return;
  const int k = kt * 32 + ((v2 >> 2) << 4) + ((lane >> 4) << 3) + (v2 & 3) * 2;
  const int n = nt * 16 + (lane & 15);
  const float* We = W + (long)e * K * N;
  S[idx] = pack2_bf16_rne(We[(long)k * N + n], We[(long)(k + 1) * N + n]);
}

// ---------------------------------------------------------------- router

__global__ __launch_bounds__(256)
void ln_router_kernel(const float* __restrict__ x, const float* __restrict__ gate_w,
                      const float* __restrict__ ln_g, const float* __restrict__ ln_b,
                      __bf16* __restrict__ xnorm, float* __restrict__ gws,
                      int* __restrict__ expid, float* __restrict__ probs8,
                      float* __restrict__ lse2) {
  __shared__ float red[256];
  __shared__ float logits_s[NE];
  const int t = blockIdx.x, tid = threadIdx.x;
  const float* xt = x + (long)t * D_MODEL;

  float v[4];
  float s = 0.f;
#pragma unroll
  for (int j = 0; j < 4; ++j) { v[j] = xt[tid + j * 256]; s += v[j]; }
  const float mu = block_reduce_sum_256(s, red) * (1.0f / D_MODEL);

  float vs = 0.f;
#pragma unroll
  for (int j = 0; j < 4; ++j) { const float d = v[j] - mu; vs += d * d; }
  const float var  = block_reduce_sum_256(vs, red) * (1.0f / D_MODEL);
  const float rstd = rsqrtf(var + 1e-5f);

  float xn[4];
#pragma unroll
  for (int j = 0; j < 4; ++j) {
    const int i = tid + j * 256;
    xn[j] = (v[j] - mu) * rstd * ln_g[i] + ln_b[i];
    xnorm[(long)t * D_MODEL + i] = (__bf16)xn[j];
  }

  for (int e = 0; e < NE; ++e) {
    float p = 0.f;
#pragma unroll
    for (int j = 0; j < 4; ++j) p += xn[j] * gate_w[e * D_MODEL + tid + j * 256];
    const float lg = block_reduce_sum_256(p, red);
    if (tid == 0) logits_s[e] = fminf(10.0f, fmaxf(-10.0f, lg));
  }
  __syncthreads();

  if (tid == 0) {
    float mx = -1e30f;
    for (int e = 0; e < NE; ++e) mx = fmaxf(mx, logits_s[e]);
    float pr[NE]; float se = 0.f;
    for (int e = 0; e < NE; ++e) { pr[e] = __expf(logits_s[e] - mx); se += pr[e]; }
    const float lse = mx + __logf(se);
    lse2[t] = lse * lse;
    int e0 = 0, e1 = 0; float p0 = -1.f, p1 = -1.f;
    const float inv = 1.0f / se;
    for (int e = 0; e < NE; ++e) {
      const float pe = pr[e] * inv;
      probs8[(long)t * NE + e] = pe;
      if (pe > p0)      { p1 = p0; e1 = e0; p0 = pe; e0 = e; }
      else if (pe > p1) { p1 = pe; e1 = e; }
    }
    const float denom = 1.0f / (p0 + p1 + 1e-8f);
    gws[t * 2 + 0] = p0 * denom;  gws[t * 2 + 1] = p1 * denom;
    expid[t * 2 + 0] = e0;        expid[t * 2 + 1] = e1;
  }
}

// ---------------------------------------------------------------- routing scatter

__global__ void init_counts_kernel(int* cnt, int* cursor) {
  const int i = threadIdx.x;
  if (i < NE) { cnt[i] = 0; cursor[i] = 0; }
}

__global__ __launch_bounds__(256)
void count_kernel(const int* __restrict__ expid, int* __restrict__ cnt) {
  const int a = blockIdx.x * 256 + threadIdx.x;
  if (a < NASSIGN) atomicAdd(&cnt[expid[a]], 1);
}

__global__ void offsets_kernel(const int* __restrict__ cnt, int* __restrict__ basep) {
  if (threadIdx.x == 0) {
    int run = 0;
    for (int e = 0; e < NE; ++e) { basep[e] = run; run += cnt[e]; }
  }
}

__global__ __launch_bounds__(256)
void scatter_kernel(const int* __restrict__ expid, const int* __restrict__ basep,
                    int* __restrict__ cursor, int* __restrict__ list,
                    int* __restrict__ slotOf) {
  const int a = blockIdx.x * 256 + threadIdx.x;
  if (a < NASSIGN) {
    const int e = expid[a];
    const int pos = basep[e] + atomicAdd(&cursor[e], 1);
    list[pos] = a >> 1;   // token id
    slotOf[a] = pos;
  }
}

// ---------------------------------------------------------------- fused expert FFN
// Block = 64 tokens of expert blockIdx.y. 512 threads = 16 waves:
// wave w -> ms = w&1 (rows ms*32..ms*32+31, two 16-row M tiles), gj = w>>1.
// GEMM1: H[64,KC] = gelu(Xn[64,1024] @ W1[:,chunk] + b1), staged bf16 in LDS.
// GEMM2: Y[64,1024] += H[64,KC] @ W2[chunk,:], accumulated in VGPRs.
// B fragments are double-buffered in registers; weight streams are prefetched
// one GEMM phase ahead with global_prefetch.

__global__ __launch_bounds__(512)
void moe_ffn_kernel(const __bf16* __restrict__ xnorm,
                    const uint4* __restrict__ w1s, const float* __restrict__ pb1,
                    const uint4* __restrict__ w2s, const float* __restrict__ pb2,
                    const int* __restrict__ cnt, const int* __restrict__ basep,
                    const int* __restrict__ list, float* __restrict__ ybuf) {
  const int e = blockIdx.y;
  const int blockStart = blockIdx.x * BM;
  const int count = cnt[e];
  if (blockStart >= count) return;

  extern __shared__ char smem[];
  __bf16* ldsA = (__bf16*)smem;                          // [BM][1024] bf16 = 128KB
  __bf16* ldsH = (__bf16*)(smem + BM * D_MODEL * 2);     // [BM][KC]   bf16 = 16KB

  const int tid = threadIdx.x;

  // Gather 64 token rows of x_norm (bf16) into LDS; pad with zeros.
  {
    const int r = tid >> 3, c8 = tid & 7;    // 8 threads/row, 16 x uint4 each
    const int gr = blockStart + r;
    uint4* dst = (uint4*)(ldsA + r * D_MODEL) + c8 * 16;
    if (gr < count) {
      const int tok = list[basep[e] + gr];
      const uint4* src = (const uint4*)(xnorm + (long)tok * D_MODEL) + c8 * 16;
#pragma unroll
      for (int q = 0; q < 16; ++q) dst[q] = src[q];
    } else {
      const uint4 z = {0u, 0u, 0u, 0u};
#pragma unroll
      for (int q = 0; q < 16; ++q) dst[q] = z;
    }
  }
  __syncthreads();

  const int wave = tid >> 5;
  const int ms   = wave & 1;     // row half: rows ms*32 .. ms*32+31
  const int gj   = wave >> 1;    // N group
  const int lane = tid & 31;
  const int nlo  = lane & 15;
  const int mofs = (lane >> 4) * 8;

  // per-expert swizzled weight bases: 8192 tiles x 64 uint4 (1KB) each
  const uint4* S1e = w1s + (long)e * 8192 * 64;   // tiles [nt(256)][kt(32)]
  const uint4* S2e = w2s + (long)e * 8192 * 64;   // tiles [nt(64)][kt(128)]

  v8f yacc0[8], yacc1[8];
#pragma unroll
  for (int t8 = 0; t8 < 8; ++t8)
#pragma unroll
    for (int q = 0; q < 8; ++q) { yacc0[t8][q] = 0.f; yacc1[t8][q] = 0.f; }

  for (int kk = 0; kk < D_FF; kk += KC) {
    // Prefetch this chunk's W2 tiles (8 x 4KB per wave), consumed in GEMM2
    // after the ~64-WMMA GEMM1 phase.
    {
      const int kt2p = kk >> 5;
#pragma unroll
      for (int tj = 0; tj < 8; ++tj) {
        const char* pf = (const char*)(S2e + ((long)(gj * 8 + tj) * 128 + kt2p) * 64);
        __builtin_prefetch(pf + lane * 128, 0, 0);
      }
    }

    // ---- GEMM1: two 16x16 H tiles: rows ms*32 and ms*32+16, cols kk+gj*16
    v8f hc0, hc1;
#pragma unroll
    for (int q = 0; q < 8; ++q) { hc0[q] = 0.f; hc1[q] = 0.f; }
    const int ncol = kk + gj * 16;
    const uint4* bp = S1e + (long)(ncol >> 4) * 32 * 64 + lane * 2;
    AFrag bcur;
    bcur.q[0] = bp[0]; bcur.q[1] = bp[1];
#pragma unroll 4
    for (int k = 0; k < D_MODEL; k += 32) {
      AFrag bnxt;
      const uint4* pn = (k + 32 < D_MODEL) ? (bp + 64) : bp;
      bnxt.q[0] = pn[0]; bnxt.q[1] = pn[1];
      const v16bf a0 = load_a_frag(ldsA, D_MODEL, ms * 32, k);
      const v16bf a1 = load_a_frag(ldsA, D_MODEL, ms * 32 + 16, k);
      hc0 = __builtin_amdgcn_wmma_f32_16x16x32_bf16(false, a0, false, bcur.v,
                                                    (short)0, hc0, false, false);
      hc1 = __builtin_amdgcn_wmma_f32_16x16x32_bf16(false, a1, false, bcur.v,
                                                    (short)0, hc1, false, false);
      bcur = bnxt;
      bp += 64;
    }

    // bias + exact GELU -> bf16 H tiles in LDS
    const float bias = pb1[e * D_FF + ncol + nlo];
    __syncthreads();   // prior chunk's GEMM2 reads of ldsH are done
#pragma unroll
    for (int q = 0; q < 8; ++q) {
      const int m = q + mofs;
      float h0 = hc0[q] + bias;
      float h1 = hc1[q] + bias;
      h0 = 0.5f * h0 * (1.0f + erff(h0 * 0.70710678118654752f));
      h1 = 0.5f * h1 * (1.0f + erff(h1 * 0.70710678118654752f));
      ldsH[(ms * 32 + m) * KC + gj * 16 + nlo]      = (__bf16)h0;
      ldsH[(ms * 32 + 16 + m) * KC + gj * 16 + nlo] = (__bf16)h1;
    }
    __syncthreads();

    // Prefetch next chunk's W1 stream (32KB contiguous per wave).
    if (kk + KC < D_FF) {
      const char* pf = (const char*)(S1e + (long)((kk + KC + gj * 16) >> 4) * 32 * 64);
#pragma unroll
      for (int r8 = 0; r8 < 8; ++r8)
        __builtin_prefetch(pf + r8 * 4096 + lane * 128, 0, 0);
    }

    // ---- GEMM2 partial: Y[rows ms*32..+32, cols gj*128..+128] += Hc @ W2[chunk]
#pragma unroll
    for (int ks = 0; ks < KC; ks += 32) {
      const v16bf a20 = load_a_frag(ldsH, KC, ms * 32, ks);
      const v16bf a21 = load_a_frag(ldsH, KC, ms * 32 + 16, ks);
      const uint4* bp2 = S2e + ((long)(gj * 8) * 128 + ((kk + ks) >> 5)) * 64 + lane * 2;
      AFrag b2cur;
      b2cur.q[0] = bp2[0]; b2cur.q[1] = bp2[1];
#pragma unroll
      for (int tj = 0; tj < 8; ++tj) {
        AFrag b2nxt;
        const uint4* pn = (tj < 7) ? (bp2 + 128 * 64) : bp2;
        b2nxt.q[0] = pn[0]; b2nxt.q[1] = pn[1];
        yacc0[tj] = __builtin_amdgcn_wmma_f32_16x16x32_bf16(false, a20, false,
                                                            b2cur.v, (short)0,
                                                            yacc0[tj], false, false);
        yacc1[tj] = __builtin_amdgcn_wmma_f32_16x16x32_bf16(false, a21, false,
                                                            b2cur.v, (short)0,
                                                            yacc1[tj], false, false);
        b2cur = b2nxt;
        bp2 += 128 * 64;
      }
    }
  }

  // ---- epilogue: + b2, store per-assignment rows into ybuf (slot-indexed)
#pragma unroll
  for (int tj = 0; tj < 8; ++tj) {
    const int col = gj * 128 + tj * 16 + nlo;
    const float bb = pb2[e * D_MODEL + col];
#pragma unroll
    for (int q = 0; q < 8; ++q) {
      const int m = q + mofs;
      const int gr0 = blockStart + ms * 32 + m;
      const int gr1 = gr0 + 16;
      if (gr0 < count)
        ybuf[(long)(basep[e] + gr0) * D_MODEL + col] = yacc0[tj][q] + bb;
      if (gr1 < count)
        ybuf[(long)(basep[e] + gr1) * D_MODEL + col] = yacc1[tj][q] + bb;
    }
  }
}

// ---------------------------------------------------------------- combine & losses

__global__ __launch_bounds__(256)
void combine_kernel(const float* __restrict__ x, const float* __restrict__ gws,
                    const int* __restrict__ slotOf, const float* __restrict__ ybuf,
                    float* __restrict__ out) {
  const int t = blockIdx.x, tid = threadIdx.x;
  const float g0 = gws[t * 2 + 0], g1 = gws[t * 2 + 1];
  const long p0 = (long)slotOf[t * 2 + 0] * D_MODEL;
  const long p1 = (long)slotOf[t * 2 + 1] * D_MODEL;
  const long xb = (long)t * D_MODEL;
#pragma unroll
  for (int j = 0; j < 4; ++j) {
    const int i = tid + j * 256;
    out[xb + i] = x[xb + i] + g0 * ybuf[p0 + i] + g1 * ybuf[p1 + i];
  }
}

__global__ __launch_bounds__(256)
void aux_kernel(const float* __restrict__ lse2, const float* __restrict__ probs8,
                const int* __restrict__ expid, float* __restrict__ out_aux) {
  __shared__ float red[256];
  const int tid = threadIdx.x;
  float zs = 0.f, ps[NE], c1[NE];
#pragma unroll
  for (int e = 0; e < NE; ++e) { ps[e] = 0.f; c1[e] = 0.f; }
  for (int t = tid; t < NTOK; t += 256) {
    zs += lse2[t];
#pragma unroll
    for (int e = 0; e < NE; ++e) ps[e] += probs8[(long)t * NE + e];
    c1[expid[t * 2]] += 1.0f;
  }
  const float zsum = block_reduce_sum_256(zs, red);
  float psum_r[NE], c1_r[NE];
  for (int e = 0; e < NE; ++e) {
    psum_r[e] = block_reduce_sum_256(ps[e], red);
    c1_r[e]   = block_reduce_sum_256(c1[e], red);
  }
  if (tid == 0) {
    const float invT = 1.0f / (float)NTOK;
    float aux = 0.f;
    for (int e = 0; e < NE; ++e) aux += (c1_r[e] * invT) * (psum_r[e] * invT);
    out_aux[0] = aux * (float)NE * 0.01f + 0.001f * (zsum * invT);
  }
}

// ---------------------------------------------------------------- launcher

extern "C" void kernel_launch(void* const* d_in, const int* in_sizes, int n_in,
                              void* d_out, int out_size, void* d_ws, size_t ws_size,
                              hipStream_t stream) {
  const float* x      = (const float*)d_in[0];
  const float* gate_w = (const float*)d_in[1];
  const float* ln_g   = (const float*)d_in[2];
  const float* ln_b   = (const float*)d_in[3];
  const float* w1     = (const float*)d_in[4];
  const float* b1     = (const float*)d_in[5];
  const float* w2     = (const float*)d_in[6];
  const float* b2     = (const float*)d_in[7];
  float* out     = (float*)d_out;
  float* out_aux = out + (size_t)NTOK * D_MODEL;

  char* ws = (char*)d_ws;
  size_t off = 0;
  auto alloc = [&](size_t bytes) -> void* {
    void* p = ws + off;
    off = (off + bytes + 255) & ~(size_t)255;
    return p;
  };
  __bf16*   xnorm  = (__bf16*)  alloc((size_t)NTOK * D_MODEL * 2);
  unsigned* w1s    = (unsigned*)alloc((size_t)NE * D_MODEL * D_FF * 2);  // 64MB
  unsigned* w2s    = (unsigned*)alloc((size_t)NE * D_FF * D_MODEL * 2);  // 64MB
  float*    gws    = (float*)   alloc((size_t)NASSIGN * 4);
  int*      expid  = (int*)     alloc((size_t)NASSIGN * 4);
  float*    probs8 = (float*)   alloc((size_t)NTOK * NE * 4);
  float*    lse2   = (float*)   alloc((size_t)NTOK * 4);
  int*      slotOf = (int*)     alloc((size_t)NASSIGN * 4);
  int*      list   = (int*)     alloc((size_t)NASSIGN * 4);
  int*      cnt    = (int*)     alloc(NE * 4);
  int*      basep  = (int*)     alloc(NE * 4);
  int*      cursor = (int*)     alloc(NE * 4);
  float*    ybuf   = (float*)   alloc((size_t)NASSIGN * D_MODEL * 4);

  // weight swizzle: fp32 -> bf16 fragments (dword count = NE*K*N/2 each)
  {
    const long dw1 = (long)NE * D_MODEL * D_FF / 2;   // 16,777,216
    swizzle_w_kernel<<<(unsigned)(dw1 / 256), 256, 0, stream>>>(w1, w1s,
                                                                D_MODEL, D_FF);
    const long dw2 = (long)NE * D_FF * D_MODEL / 2;
    swizzle_w_kernel<<<(unsigned)(dw2 / 256), 256, 0, stream>>>(w2, w2s,
                                                                D_FF, D_MODEL);
  }

  ln_router_kernel<<<NTOK, 256, 0, stream>>>(x, gate_w, ln_g, ln_b,
                                             xnorm, gws, expid, probs8, lse2);
  init_counts_kernel<<<1, 32, 0, stream>>>(cnt, cursor);
  count_kernel<<<NASSIGN / 256, 256, 0, stream>>>(expid, cnt);
  offsets_kernel<<<1, 1, 0, stream>>>(cnt, basep);
  scatter_kernel<<<NASSIGN / 256, 256, 0, stream>>>(expid, basep, cursor,
                                                    list, slotOf);
  const size_t lds_bytes = (size_t)BM * D_MODEL * 2 + (size_t)BM * KC * 2; // 144KB
  moe_ffn_kernel<<<dim3(NTOK / BM, NE), 512, lds_bytes, stream>>>(
      xnorm, (const uint4*)w1s, b1, (const uint4*)w2s, b2,
      cnt, basep, list, ybuf);
  combine_kernel<<<NTOK, 256, 0, stream>>>(x, gws, slotOf, ybuf, out);
  aux_kernel<<<1, 256, 0, stream>>>(lse2, probs8, expid, out_aux);
}
